// DFA_63720134804167
// MI455X (gfx1250) — matrix operand order repaired and checked by reading, not compiled, not run on hardware
//
#include <hip/hip_runtime.h>
#include <hip/hip_bf16.h>
#include <cstdint>
#include <cstddef>

// ---- CDNA5 (gfx1250) wave32 WMMA types ----
typedef __bf16    v16bf __attribute__((ext_vector_type(16)));
typedef float     v8f   __attribute__((ext_vector_type(8)));
typedef unsigned  v4u   __attribute__((ext_vector_type(4)));

union FragBF { unsigned u[8]; v4u q[2]; v16bf v; };  // all trivially-copyable

#define HH  200
#define WW  200
#define LQ  40000
#define CC  512
#define NSA 96     // 64 offset channels + 32 attn-logit channels

__device__ __forceinline__ unsigned short f2bf(float f) {
  union { float f; unsigned u; } v; v.f = f;
  unsigned r = v.u + 0x7FFFu + ((v.u >> 16) & 1u);   // round-to-nearest-even
  return (unsigned short)(r >> 16);
}

__device__ __forceinline__ v8f wmma_bf16(const FragBF& a, const FragBF& b,
                                         v8f c) {
  return __builtin_amdgcn_wmma_f32_16x16x32_bf16(
      false, a.v, false, b.v, (short)0, c, false, false);
}

// ---------------------------------------------------------------------------
// Kernel 1: convert weights to bf16; fuse w_samp(512x64)+w_attn(512x32) into
// one 512x96 matrix so offsets+attn become a single WMMA GEMM.
// ---------------------------------------------------------------------------
__global__ void DFA_prep_weights(const float* __restrict__ wv,
                                 const float* __restrict__ wo,
                                 const float* __restrict__ wsmp,
                                 const float* __restrict__ watt,
                                 unsigned short* __restrict__ wvBf,
                                 unsigned short* __restrict__ woBf,
                                 unsigned short* __restrict__ wsaBf) {
  int t = blockIdx.x * 256 + threadIdx.x;          // 2240*256 == 573440 exactly
  if (t < 262144) {
    wvBf[t] = f2bf(wv[t]);
  } else if (t < 524288) {
    int i = t - 262144;
    woBf[i] = f2bf(wo[i]);
  } else {
    int i = t - 524288;                            // 0..49151
    int k = i / NSA, n = i % NSA;
    float x = (n < 64) ? wsmp[k * 64 + n] : watt[k * 32 + (n - 64)];
    wsaBf[i] = f2bf(x);
  }
}

// ---------------------------------------------------------------------------
// Kernel 2: transpose bev (C,Lq) -> row-major bf16 src[q][c] and
// query[q][c] = src + learned pos, via 64x64 LDS tile (both sides coalesced).
// pos[h*W+w][c] = c<256 ? row_embed[w][c] : col_embed[h][c-256]
// ---------------------------------------------------------------------------
__global__ void DFA_prep_srcq(const float* __restrict__ bev,
                              const float* __restrict__ colE,
                              const float* __restrict__ rowE,
                              unsigned short* __restrict__ srcBf,
                              unsigned short* __restrict__ qryBf) {
  __shared__ float lds[64 * 65];                   // +1 pad: conflict-free cols
  const int t = threadIdx.x;
  const int qBase = blockIdx.x * 64;               // 625 tiles along Lq
  const int cBase = blockIdx.y * 64;               // 8 tiles along C
  const int qi = t & 63, ci = t >> 6;
#pragma unroll
  for (int i = 0; i < 16; ++i) {
    int cL = ci + i * 4;
    lds[cL * 65 + qi] = bev[(size_t)(cBase + cL) * LQ + (qBase + qi)];
  }
  __syncthreads();
#pragma unroll
  for (int i = 0; i < 16; ++i) {
    int qL = ci + i * 4;
    int q  = qBase + qL;
    int c  = cBase + qi;
    float f = lds[qi * 65 + qL];
    int h = q / WW, w = q % WW;
    float pos = (c < 256) ? rowE[w * 256 + c] : colE[h * 256 + (c - 256)];
    srcBf[(size_t)q * CC + c] = f2bf(f);
    qryBf[(size_t)q * CC + c] = f2bf(f + pos);
  }
}

// ---------------------------------------------------------------------------
// A-fragment loader: 16x32 bf16 tile, CDNA5 layout.
// lane<16: row M=lane, K pairs {0..7} in v0..3, {16..23} in v4..7
// lane>=16: same rows, K pairs shifted by +8 / +24.
// (compiler merges the 4 consecutive pair-loads into b128s)
// ---------------------------------------------------------------------------
__device__ __forceinline__ void loadA(FragBF& a, const unsigned short* Arow,
                                      int kBase) {
#pragma unroll
  for (int v = 0; v < 8; ++v) {
    int k = kBase + (v & 3) * 2 + ((v & 4) ? 16 : 0);
    a.u[v] = *(const unsigned*)(Arow + k);
  }
}

// ---------------------------------------------------------------------------
// Kernel 3: value = src @ w_val + b_val   (M=40000, N=512, K=512), f32 out.
// Block = 8 waves; each wave owns a 32x64 tile: 2 A fragments x 4 N tiles =
// 8 accumulators. All 4 B fragments are pre-loaded into distinct registers
// so the scheduler can stagger s_wait_loadcnt instead of draining to 0, and
// each B fragment feeds two independent WMMAs.
// ---------------------------------------------------------------------------
__global__ void DFA_gemm_value(const unsigned short* __restrict__ A,
                               const unsigned short* __restrict__ Bw,
                               const float* __restrict__ bias,
                               float* __restrict__ D) {
  const int wave = threadIdx.x >> 5, lane = threadIdx.x & 31;
  const int mBase = blockIdx.x * 32;               // 1250 blocks
  const int nBase = wave * 64;
  const int nLane = lane & 15, mSel = lane >> 4;
  v8f acc0[4], acc1[4];
#pragma unroll
  for (int t = 0; t < 4; ++t) {
    float bv = bias[nBase + t * 16 + nLane];
#pragma unroll
    for (int j = 0; j < 8; ++j) { acc0[t][j] = bv; acc1[t][j] = bv; }
  }
  const unsigned short* Arow0 = A + (size_t)(mBase + nLane) * CC + mSel * 8;
  const unsigned short* Arow1 = Arow0 + 16 * CC;
  for (int kb = 0; kb < 16; ++kb) {
    const int kBase = kb * 32;
    FragBF a0, a1;
    loadA(a0, Arow0, kBase);
    loadA(a1, Arow1, kBase);
    const unsigned short* Brow = Bw + (size_t)(kBase + mSel * 8) * CC + nBase + nLane;
    if (kb < 15) __builtin_prefetch(Brow + 32 * CC, 0, 1);   // next K-slab rows
    FragBF b[4];
#pragma unroll
    for (int t = 0; t < 4; ++t) {
      for (int v = 0; v < 8; ++v) {
        int k = (v & 3) * 2 + ((v & 4) ? 16 : 0);
        b[t].u[v] = (unsigned)Brow[k * CC + t * 16] |
                    ((unsigned)Brow[(k + 1) * CC + t * 16] << 16);
      }
    }
#pragma unroll
    for (int t = 0; t < 4; ++t) {
      acc0[t] = wmma_bf16(a0, b[t], acc0[t]);
      acc1[t] = wmma_bf16(a1, b[t], acc1[t]);
    }
  }
#pragma unroll
  for (int t = 0; t < 4; ++t)
#pragma unroll
    for (int j = 0; j < 8; ++j) {
      int n = nBase + t * 16 + nLane;
      D[(size_t)(mBase + mSel * 8 + j) * CC + n]      = acc0[t][j];
      D[(size_t)(mBase + 16 + mSel * 8 + j) * CC + n] = acc1[t][j];
    }
}

// ---------------------------------------------------------------------------
// Kernel 4: [offsets|attn_logits] = query @ w_sa + bias  (M=40000, N=96).
// One wave per 16x96 tile (6 accumulators).
// ---------------------------------------------------------------------------
__global__ void DFA_gemm_sa(const unsigned short* __restrict__ A,
                            const unsigned short* __restrict__ Bw,
                            const float* __restrict__ bSamp,
                            const float* __restrict__ bAttn,
                            float* __restrict__ D) {
  const int wave = threadIdx.x >> 5, lane = threadIdx.x & 31;
  const int mTile = blockIdx.x * 8 + wave;
  if (mTile >= 2500) return;
  const int mBase = mTile * 16;
  const int nLane = lane & 15, mSel = lane >> 4;
  v8f acc[6];
#pragma unroll
  for (int t = 0; t < 6; ++t) {
    int n = t * 16 + nLane;
    float bv = (n < 64) ? bSamp[n] : bAttn[n - 64];
#pragma unroll
    for (int j = 0; j < 8; ++j) acc[t][j] = bv;
  }
  const unsigned short* Arow = A + (size_t)(mBase + nLane) * CC + mSel * 8;
  for (int kb = 0; kb < 16; ++kb) {
    const int kBase = kb * 32;
    FragBF a; loadA(a, Arow, kBase);
    const unsigned short* Brow = Bw + (size_t)(kBase + mSel * 8) * NSA + nLane;
    FragBF b[6];
#pragma unroll
    for (int t = 0; t < 6; ++t) {
      for (int v = 0; v < 8; ++v) {
        int k = (v & 3) * 2 + ((v & 4) ? 16 : 0);
        b[t].u[v] = (unsigned)Brow[k * NSA + t * 16] |
                    ((unsigned)Brow[(k + 1) * NSA + t * 16] << 16);
      }
    }
#pragma unroll
    for (int t = 0; t < 6; ++t) acc[t] = wmma_bf16(a, b[t], acc[t]);
  }
#pragma unroll
  for (int t = 0; t < 6; ++t)
#pragma unroll
    for (int j = 0; j < 8; ++j)
      D[(size_t)(mBase + mSel * 8 + j) * NSA + t * 16 + nLane] = acc[t][j];
}

// ---------------------------------------------------------------------------
// Kernel 5: per-(q,h) softmax + bilinear sample + weighted reduce.
// Block = one query (8 waves = 8 heads); lane owns 2 of the 64 head channels
// so each corner gather is a coalesced 256B float2 row-fetch from `value`.
// hidden is written in bf16 for the final WMMA GEMM.
// ---------------------------------------------------------------------------
__global__ void DFA_sample(const float* __restrict__ value,
                           const float* __restrict__ sa,
                           unsigned short* __restrict__ hidden) {
  const int q = blockIdx.x;
  const int h = threadIdx.x >> 5, lane = threadIdx.x & 31;
  const float* sq = sa + (size_t)q * NSA;
  float lg[4];
#pragma unroll
  for (int p = 0; p < 4; ++p) lg[p] = sq[64 + h * 4 + p];
  float mx = fmaxf(fmaxf(lg[0], lg[1]), fmaxf(lg[2], lg[3]));
  float ex[4]; float sum = 0.f;
#pragma unroll
  for (int p = 0; p < 4; ++p) { ex[p] = __expf(lg[p] - mx); sum += ex[p]; }
  const float inv = 1.f / sum;
  const float bx = (float)(q % WW) - 0.5f;   // ref_x*W - 0.5
  const float by = (float)(q / WW) - 0.5f;   // ref_y*H - 0.5
  const float* vcol = value + h * 64 + lane * 2;
  float a0 = 0.f, a1 = 0.f;
#pragma unroll
  for (int p = 0; p < 4; ++p) {
    float x = bx + sq[h * 8 + p * 2 + 0];    // offsets/W * W == offsets
    float y = by + sq[h * 8 + p * 2 + 1];
    float fx = floorf(x), fy = floorf(y);
    float tx = x - fx, ty = y - fy;
    int x0 = (int)fx, y0 = (int)fy;
    float ap = ex[p] * inv;
#pragma unroll
    for (int dy = 0; dy < 2; ++dy)
#pragma unroll
      for (int dx = 0; dx < 2; ++dx) {
        int xi = x0 + dx, yi = y0 + dy;
        bool valid = (xi >= 0) & (xi < WW) & (yi >= 0) & (yi < HH);
        int xc = min(max(xi, 0), WW - 1);
        int yc = min(max(yi, 0), HH - 1);
        float wgt = valid ? (dx ? tx : 1.f - tx) * (dy ? ty : 1.f - ty) * ap
                          : 0.f;
        float2 v = *(const float2*)(vcol + (size_t)(yc * WW + xc) * CC);
        a0 += wgt * v.x; a1 += wgt * v.y;
      }
  }
  unsigned pk = (unsigned)f2bf(a0) | ((unsigned)f2bf(a1) << 16);
  *(unsigned*)(hidden + (size_t)q * CC + h * 64 + lane * 2) = pk;
}

// ---------------------------------------------------------------------------
// Kernel 6: out = hidden @ w_out + b_out, stored TRANSPOSED to (C,H,W).
// Same 32x64 WMMA tiling as kernel 3; each wave transposes its tile through
// a private 16x68 LDS region in two passes (DS ops are in-order within a
// wave, so the WAR between passes is safe and no barrier is needed), then
// emits 128B-contiguous float4 runs along Lq.
// ---------------------------------------------------------------------------
__global__ void DFA_gemm_out(const unsigned short* __restrict__ A,
                             const unsigned short* __restrict__ Bw,
                             const float* __restrict__ bias,
                             float* __restrict__ D /* [C][LQ] */) {
  __shared__ float tileLds[8][16 * 68];            // per-wave 16x64 (+pad)
  const int wave = threadIdx.x >> 5, lane = threadIdx.x & 31;
  const int mBase = blockIdx.x * 32;               // 1250 blocks
  const int nBase = wave * 64;
  const int nLane = lane & 15, mSel = lane >> 4;
  v8f acc0[4], acc1[4];
#pragma unroll
  for (int t = 0; t < 4; ++t) {
    float bv = bias[nBase + t * 16 + nLane];
#pragma unroll
    for (int j = 0; j < 8; ++j) { acc0[t][j] = bv; acc1[t][j] = bv; }
  }
  const unsigned short* Arow0 = A + (size_t)(mBase + nLane) * CC + mSel * 8;
  const unsigned short* Arow1 = Arow0 + 16 * CC;
  for (int kb = 0; kb < 16; ++kb) {
    const int kBase = kb * 32;
    FragBF a0, a1;
    loadA(a0, Arow0, kBase);
    loadA(a1, Arow1, kBase);
    const unsigned short* Brow = Bw + (size_t)(kBase + mSel * 8) * CC + nBase + nLane;
    if (kb < 15) __builtin_prefetch(Brow + 32 * CC, 0, 1);
    FragBF b[4];
#pragma unroll
    for (int t = 0; t < 4; ++t) {
      for (int v = 0; v < 8; ++v) {
        int k = (v & 3) * 2 + ((v & 4) ? 16 : 0);
        b[t].u[v] = (unsigned)Brow[k * CC + t * 16] |
                    ((unsigned)Brow[(k + 1) * CC + t * 16] << 16);
      }
    }
#pragma unroll
    for (int t = 0; t < 4; ++t) {
      acc0[t] = wmma_bf16(a0, b[t], acc0[t]);
      acc1[t] = wmma_bf16(a1, b[t], acc1[t]);
    }
  }
  float* tl = tileLds[wave];
#pragma unroll
  for (int half = 0; half < 2; ++half) {
    // pass: dump 16xM half-tile to LDS, then stream out columns
#pragma unroll
    for (int t = 0; t < 4; ++t)
#pragma unroll
      for (int j = 0; j < 8; ++j)
        tl[(mSel * 8 + j) * 68 + t * 16 + nLane] =
            half ? acc1[t][j] : acc0[t][j];
    // same-wave DS ordering: stores above complete before loads below
#pragma unroll
    for (int it = 0; it < 2; ++it) {
      int n = it * 32 + lane;                      // local column 0..63
      float col[16];
#pragma unroll
      for (int m = 0; m < 16; ++m) col[m] = tl[m * 68 + n];
      float* dst = D + (size_t)(nBase + n) * LQ + mBase + half * 16;
      *(float4*)(dst + 0)  = make_float4(col[0],  col[1],  col[2],  col[3]);
      *(float4*)(dst + 4)  = make_float4(col[4],  col[5],  col[6],  col[7]);
      *(float4*)(dst + 8)  = make_float4(col[8],  col[9],  col[10], col[11]);
      *(float4*)(dst + 12) = make_float4(col[12], col[13], col[14], col[15]);
    }
  }
}

// ---------------------------------------------------------------------------
extern "C" void kernel_launch(void* const* d_in, const int* in_sizes, int n_in,
                              void* d_out, int out_size, void* d_ws,
                              size_t ws_size, hipStream_t stream) {
  (void)in_sizes; (void)n_in; (void)out_size; (void)ws_size;
  const float* bev    = (const float*)d_in[0];
  const float* colE   = (const float*)d_in[1];
  const float* rowE   = (const float*)d_in[2];
  const float* w_val  = (const float*)d_in[3];
  const float* b_val  = (const float*)d_in[4];
  const float* w_samp = (const float*)d_in[5];
  const float* b_samp = (const float*)d_in[6];
  const float* w_attn = (const float*)d_in[7];
  const float* b_attn = (const float*)d_in[8];
  const float* w_out  = (const float*)d_in[9];
  const float* b_out  = (const float*)d_in[10];

  char* ws = (char*)d_ws;                       // total ~221.3 MB
  unsigned short* srcBf  = (unsigned short*)(ws + 0);           // 40.96 MB
  unsigned short* qryBf  = (unsigned short*)(ws + 40960000);    // 40.96 MB
  unsigned short* wvalBf = (unsigned short*)(ws + 81920000);    // 0.52 MB
  unsigned short* woutBf = (unsigned short*)(ws + 82444288);    // 0.52 MB
  unsigned short* wsaBf  = (unsigned short*)(ws + 82968576);    // 0.10 MB
  float*          value  = (float*)(ws + 83066880);             // 81.92 MB
  float*          saBuf  = (float*)(ws + 164986880);            // 15.36 MB
  unsigned short* hidden = (unsigned short*)(ws + 180346880);   // 40.96 MB

  DFA_prep_weights<<<2240, 256, 0, stream>>>(w_val, w_out, w_samp, w_attn,
                                             wvalBf, woutBf, wsaBf);
  DFA_prep_srcq<<<dim3(625, 8), 256, 0, stream>>>(bev, colE, rowE,
                                                  srcBf, qryBf);
  DFA_gemm_value<<<1250, 256, 0, stream>>>(srcBf, wvalBf, b_val, value);
  DFA_gemm_sa<<<313, 256, 0, stream>>>(qryBf, wsaBf, b_samp, b_attn, saBuf);
  DFA_sample<<<40000, 256, 0, stream>>>(value, saBuf, hidden);
  DFA_gemm_out<<<1250, 256, 0, stream>>>(hidden, woutBf, b_out, (float*)d_out);
}
